// ObjectClassifier_33105607918058
// MI455X (gfx1250) — matrix operand order repaired and checked by reading, not compiled, not run on hardware
//
#include <hip/hip_runtime.h>
#include <hip/hip_bf16.h>

typedef __attribute__((ext_vector_type(16))) __bf16 v16bf;
typedef __attribute__((ext_vector_type(8)))  float  v8f;

#define NROWS   32768
#define DFEAT   2048
#define NOBJ    36
#define NCLS    37
#define NSEG    64
#define KIN     2376
#define KPAD    2400
#define H1      1024
#define NC2PAD  48
#define OUT_SCORES (NROWS*36)
#define OUT_LABELS (NROWS*37)

__device__ __forceinline__ unsigned short f2bf(float f) {
    unsigned u = __float_as_uint(f);
    u += 0x7FFFu + ((u >> 16) & 1u);           // round-to-nearest-even
    return (unsigned short)(u >> 16);
}

// ---------------- small prep kernels ----------------

__global__ void k_cs_stats(const float* __restrict__ boxes, float* __restrict__ csStats) {
    __shared__ float sh[8][256];
    float s[4] = {0,0,0,0}, q[4] = {0,0,0,0};
    for (int r = threadIdx.x; r < NROWS; r += 256) {
        float b0 = boxes[4*r+0], b1 = boxes[4*r+1], b2 = boxes[4*r+2], b3 = boxes[4*r+3];
        float c[4] = { (b0+b2)*0.5f, (b1+b3)*0.5f, b2-b0, b3-b1 };
        #pragma unroll
        for (int i = 0; i < 4; ++i) { s[i] += c[i]; q[i] += c[i]*c[i]; }
    }
    #pragma unroll
    for (int i = 0; i < 4; ++i) { sh[i][threadIdx.x] = s[i]; sh[4+i][threadIdx.x] = q[i]; }
    __syncthreads();
    for (int off = 128; off > 0; off >>= 1) {
        if (threadIdx.x < off)
            #pragma unroll
            for (int i = 0; i < 8; ++i) sh[i][threadIdx.x] += sh[i][threadIdx.x + off];
        __syncthreads();
    }
    if (threadIdx.x < 4) {
        float m   = sh[threadIdx.x][0] / (float)NROWS;
        float var = sh[4+threadIdx.x][0] / (float)NROWS - m*m;
        csStats[threadIdx.x]     = m;
        csStats[4 + threadIdx.x] = rsqrtf(var + 1e-5f);
    }
}

// W1 (2376x1024 f32) -> W1T (1024 x 2400 bf16), transposed + K padded with zeros
__global__ __launch_bounds__(256) void k_cvt_w1(const float* __restrict__ w,
                                                unsigned short* __restrict__ wT) {
    __shared__ unsigned short tile[32][33];
    int kb = blockIdx.x * 32, cb = blockIdx.y * 32;
    int tx = threadIdx.x & 31, ty = threadIdx.x >> 5;   // 32 x 8
    #pragma unroll
    for (int i = 0; i < 4; ++i) {
        int k = kb + ty + i*8;
        float v = (k < KIN) ? w[(size_t)k*H1 + (cb + tx)] : 0.f;
        tile[ty + i*8][tx] = f2bf(v);
    }
    __syncthreads();
    #pragma unroll
    for (int i = 0; i < 4; ++i) {
        int c = cb + ty + i*8;
        wT[(size_t)c*KPAD + (kb + tx)] = tile[tx][ty + i*8];
    }
}

// W2 (1024x37 f32) -> W2T (48 x 1024 bf16), transposed + N padded
__global__ void k_cvt_w2(const float* __restrict__ w, unsigned short* __restrict__ wT) {
    int i = blockIdx.x * 256 + threadIdx.x;
    if (i >= NC2PAD * H1) return;
    int c = i >> 10, k = i & (H1 - 1);
    wT[i] = (c < NCLS) ? f2bf(w[(size_t)k*NCLS + c]) : (unsigned short)0;
}

// Build X = [features | dist@objW | relu(bn(cs)@posW + posB) | 0pad] in bf16, stride KPAD
__global__ __launch_bounds__(256) void k_build_x(
    const float* __restrict__ dist, const float* __restrict__ boxes,
    const float* __restrict__ feat, const float* __restrict__ objW,
    const float* __restrict__ posW, const float* __restrict__ posB,
    const float* __restrict__ bn4g, const float* __restrict__ bn4b,
    const float* __restrict__ csStats, unsigned short* __restrict__ X)
{
    __shared__ float sObjW[NOBJ * 200];   // 28.8 KB
    __shared__ float sPosW[4 * 128];
    __shared__ float sDist[8][NOBJ];
    __shared__ float sCs[8][4];
    int tid = threadIdx.x;
    for (int i = tid; i < NOBJ*200; i += 256) sObjW[i] = objW[i];
    for (int i = tid; i < 512;      i += 256) sPosW[i] = posW[i];
    int rowBase = blockIdx.x * 8;
    for (int i = tid; i < 8*NOBJ; i += 256) {
        int r = i / NOBJ, c = i % NOBJ;
        sDist[r][c] = dist[(size_t)(rowBase + r)*NOBJ + c];
    }
    if (tid < 8) {
        int r = rowBase + tid;
        float b0 = boxes[4*r+0], b1 = boxes[4*r+1], b2 = boxes[4*r+2], b3 = boxes[4*r+3];
        float c[4] = { (b0+b2)*0.5f, (b1+b3)*0.5f, b2-b0, b3-b1 };
        #pragma unroll
        for (int i = 0; i < 4; ++i)
            sCs[tid][i] = (c[i] - csStats[i]) * csStats[4+i] * bn4g[i] + bn4b[i];
    }
    __syncthreads();
    for (int r = 0; r < 8; ++r) {
        size_t grow = rowBase + r;
        for (int c = tid; c < KPAD; c += 256) {
            float v;
            if (c < DFEAT) {
                v = feat[grow*DFEAT + c];
            } else if (c < DFEAT + 200) {
                int j = c - DFEAT; float acc = 0.f;
                #pragma unroll
                for (int k = 0; k < NOBJ; ++k) acc += sDist[r][k] * sObjW[k*200 + j];
                v = acc;
            } else if (c < KIN) {
                int j = c - (DFEAT + 200); float acc = posB[j];
                #pragma unroll
                for (int k = 0; k < 4; ++k) acc += sCs[r][k] * sPosW[k*128 + j];
                v = fmaxf(acc, 0.f);
            } else v = 0.f;
            X[grow*KPAD + c] = f2bf(v);
        }
    }
}

// ---------------- WMMA fragment helpers ----------------
// A (16x32 bf16): lane<16 -> row=lane, elems 0..15 = K0..15
//                 lane>=16 -> row=lane-16, elems 0..15 = K16..31
__device__ __forceinline__ v16bf load_a(const unsigned short* rowPtr, int kb0) {
    v16bf a;
    const __bf16* p = (const __bf16*)rowPtr;
    #pragma unroll
    for (int i = 0; i < 16; ++i) a[i] = p[kb0 + i];
    return a;
}
// B (32x16 bf16) from LDS [col][k]: lane<16 -> K0..15, lane>=16 -> K16..31, n=lane%16
__device__ __forceinline__ v16bf load_b(const unsigned short* colPtr, int kb) {
    v16bf b;
    const __bf16* p = (const __bf16*)colPtr;
    #pragma unroll
    for (int i = 0; i < 16; ++i) b[i] = p[kb + i];
    return b;
}
__device__ __forceinline__ v8f wmma_bf16(v16bf a, v16bf b, v8f c) {
    return __builtin_amdgcn_wmma_f32_16x16x32_bf16(false, a, false, b, (short)0, c, false, false);
}

// ---------------- GEMM1: Y(N x 1024) = X(N x 2400) @ W1T^T + b1 ----------------
// block tile 128x128, 8 waves: wave = (wm 0..3 row-group of 32) x (wn 0..1 col-group of 64)
// double-buffered LDS, uint4 global loads, b128 LDS stores.
__global__ __launch_bounds__(256) void k_gemm1(
    const unsigned short* __restrict__ Xg, const unsigned short* __restrict__ WT,
    const float* __restrict__ bias, float* __restrict__ Y)
{
    __shared__ unsigned short Xs[2][128][32];   // 16 KB
    __shared__ unsigned short Ws[2][128][32];   // 16 KB  ([col][k])
    const int tid  = threadIdx.x;
    const int lane = tid & 31;
    const int wave = tid >> 5;
    const int wm   = wave & 3;
    const int wn   = wave >> 2;
    const int rowBase = blockIdx.y * 128;
    const int colBase = blockIdx.x * 128;
    const uint4* X16 = (const uint4*)Xg;   // row stride KPAD/8 = 300 uint4
    const uint4* W16 = (const uint4*)WT;

    const int r0 = tid >> 2, c4 = tid & 3;   // uint4 tile coords; second half row = r0+64
    const int akb  = (lane < 16) ? 0 : 16;
    const int bkb  = (lane < 16) ? 0 : 16;
    const int arow = lane & 15;
    const int bcol = lane & 15;

    const int KITER = KPAD / 32;   // 75
    uint4 xA, xB, wA, wB;

    // preload tile 0
    xA = X16[(size_t)(rowBase + r0     )*(KPAD/8) + c4];
    xB = X16[(size_t)(rowBase + r0 + 64)*(KPAD/8) + c4];
    wA = W16[(size_t)(colBase + r0     )*(KPAD/8) + c4];
    wB = W16[(size_t)(colBase + r0 + 64)*(KPAD/8) + c4];
    *(uint4*)&Xs[0][r0     ][c4*8] = xA;
    *(uint4*)&Xs[0][r0 + 64][c4*8] = xB;
    *(uint4*)&Ws[0][r0     ][c4*8] = wA;
    *(uint4*)&Ws[0][r0 + 64][c4*8] = wB;
    __syncthreads();

    v8f acc[2][4] = {};
    for (int i = 0; i < KITER; ++i) {
        const int cur = i & 1;
        if (i + 1 < KITER) {
            const int ko = (i + 1) * 4;   // 32 halves = 4 uint4
            xA = X16[(size_t)(rowBase + r0     )*(KPAD/8) + ko + c4];
            xB = X16[(size_t)(rowBase + r0 + 64)*(KPAD/8) + ko + c4];
            wA = W16[(size_t)(colBase + r0     )*(KPAD/8) + ko + c4];
            wB = W16[(size_t)(colBase + r0 + 64)*(KPAD/8) + ko + c4];
        }
        v16bf a0 = load_a(&Xs[cur][wm*32 +      arow][0], akb);
        v16bf a1 = load_a(&Xs[cur][wm*32 + 16 + arow][0], akb);
        #pragma unroll
        for (int nt = 0; nt < 4; ++nt) {
            v16bf b = load_b(&Ws[cur][wn*64 + nt*16 + bcol][0], bkb);
            acc[0][nt] = wmma_bf16(a0, b, acc[0][nt]);
            acc[1][nt] = wmma_bf16(a1, b, acc[1][nt]);
        }
        if (i + 1 < KITER) {
            const int nxt = cur ^ 1;
            *(uint4*)&Xs[nxt][r0     ][c4*8] = xA;
            *(uint4*)&Xs[nxt][r0 + 64][c4*8] = xB;
            *(uint4*)&Ws[nxt][r0     ][c4*8] = wA;
            *(uint4*)&Ws[nxt][r0 + 64][c4*8] = wB;
        }
        __syncthreads();
    }

    const int moff = (lane < 16) ? 0 : 8;
    #pragma unroll
    for (int mt = 0; mt < 2; ++mt)
        #pragma unroll
        for (int nt = 0; nt < 4; ++nt) {
            int col = colBase + wn*64 + nt*16 + bcol;
            float bv = bias[col];
            #pragma unroll
            for (int v = 0; v < 8; ++v) {
                int row = rowBase + wm*32 + mt*16 + moff + v;
                Y[(size_t)row*H1 + col] = acc[mt][nt][v] + bv;
            }
        }
}

// ---------------- bn1 stats (deterministic two-stage) ----------------
__global__ void k_colstats(const float* __restrict__ Y,
                           float* __restrict__ pSum, float* __restrict__ pSq) {
    int col = blockIdx.x * 256 + threadIdx.x;      // 0..1023
    int ck  = blockIdx.y;                          // 0..63, 512 rows each
    float s = 0.f, q = 0.f;
    for (int r = ck*512; r < ck*512 + 512; ++r) {
        float v = Y[(size_t)r*H1 + col];
        s += v; q += v*v;
    }
    pSum[ck*H1 + col] = s;
    pSq [ck*H1 + col] = q;
}

__global__ void k_colstats_final(const float* __restrict__ pSum, const float* __restrict__ pSq,
                                 float* __restrict__ bnStats) {
    int col = blockIdx.x * 256 + threadIdx.x;
    float s = 0.f, q = 0.f;
    for (int c = 0; c < 64; ++c) { s += pSum[c*H1 + col]; q += pSq[c*H1 + col]; }
    float m   = s / (float)NROWS;
    float var = q / (float)NROWS - m*m;
    bnStats[col]      = m;
    bnStats[H1 + col] = rsqrtf(var + 1e-5f);
}

__global__ void k_norm_h(const float* __restrict__ Y, const float* __restrict__ bnStats,
                         const float* __restrict__ g, const float* __restrict__ b,
                         unsigned short* __restrict__ H) {
    size_t i4 = (size_t)blockIdx.x * 256 + threadIdx.x;   // over N*H1/4
    float4 y = ((const float4*)Y)[i4];
    int col = (int)((i4 * 4) & (H1 - 1));
    float v0 = (y.x - bnStats[col+0]) * bnStats[H1+col+0] * g[col+0] + b[col+0];
    float v1 = (y.y - bnStats[col+1]) * bnStats[H1+col+1] * g[col+1] + b[col+1];
    float v2 = (y.z - bnStats[col+2]) * bnStats[H1+col+2] * g[col+2] + b[col+2];
    float v3 = (y.w - bnStats[col+3]) * bnStats[H1+col+3] * g[col+3] + b[col+3];
    uint2 o;
    o.x = (unsigned)f2bf(fmaxf(v0,0.f)) | ((unsigned)f2bf(fmaxf(v1,0.f)) << 16);
    o.y = (unsigned)f2bf(fmaxf(v2,0.f)) | ((unsigned)f2bf(fmaxf(v3,0.f)) << 16);
    ((uint2*)H)[i4] = o;
}

// ---------------- GEMM2: L(N x 48) = H(N x 1024) @ W2T^T + b2 ----------------
__global__ __launch_bounds__(256) void k_gemm2(
    const unsigned short* __restrict__ Hg, const unsigned short* __restrict__ WT,
    const float* __restrict__ bias, float* __restrict__ L)
{
    __shared__ unsigned short Hs[2][128][32];      // 16 KB
    __shared__ unsigned short Ws[2][NC2PAD][32];   // 6 KB ([col][k])
    const int tid  = threadIdx.x;
    const int lane = tid & 31;
    const int wave = tid >> 5;                 // each wave owns 16 rows
    const int rowBase = blockIdx.x * 128;
    const uint4* H16 = (const uint4*)Hg;   // row stride H1/8 = 128 uint4
    const uint4* W16 = (const uint4*)WT;

    const int r0 = tid >> 2, c4 = tid & 3;
    const int wr = tid >> 2;                   // W2 tile: threads 0..191 -> 48 cols x 4
    const int akb  = (lane < 16) ? 0 : 16;
    const int bkb  = (lane < 16) ? 0 : 16;
    const int arow = lane & 15;
    const int bcol = lane & 15;

    const int KITER = H1 / 32;   // 32
    uint4 hA, hB, w0;

    hA = H16[(size_t)(rowBase + r0     )*(H1/8) + c4];
    hB = H16[(size_t)(rowBase + r0 + 64)*(H1/8) + c4];
    if (tid < 192) w0 = W16[(size_t)wr*(H1/8) + c4];
    *(uint4*)&Hs[0][r0     ][c4*8] = hA;
    *(uint4*)&Hs[0][r0 + 64][c4*8] = hB;
    if (tid < 192) *(uint4*)&Ws[0][wr][c4*8] = w0;
    __syncthreads();

    v8f acc[3] = {};
    for (int i = 0; i < KITER; ++i) {
        const int cur = i & 1;
        if (i + 1 < KITER) {
            const int ko = (i + 1) * 4;
            hA = H16[(size_t)(rowBase + r0     )*(H1/8) + ko + c4];
            hB = H16[(size_t)(rowBase + r0 + 64)*(H1/8) + ko + c4];
            if (tid < 192) w0 = W16[(size_t)wr*(H1/8) + ko + c4];
        }
        v16bf a = load_a(&Hs[cur][wave*16 + arow][0], akb);
        #pragma unroll
        for (int nt = 0; nt < 3; ++nt) {
            v16bf b = load_b(&Ws[cur][nt*16 + bcol][0], bkb);
            acc[nt] = wmma_bf16(a, b, acc[nt]);
        }
        if (i + 1 < KITER) {
            const int nxt = cur ^ 1;
            *(uint4*)&Hs[nxt][r0     ][c4*8] = hA;
            *(uint4*)&Hs[nxt][r0 + 64][c4*8] = hB;
            if (tid < 192) *(uint4*)&Ws[nxt][wr][c4*8] = w0;
        }
        __syncthreads();
    }

    const int moff = (lane < 16) ? 0 : 8;
    #pragma unroll
    for (int nt = 0; nt < 3; ++nt) {
        int col = nt*16 + bcol;
        float bv = (col < NCLS) ? bias[col] : 0.f;
        #pragma unroll
        for (int v = 0; v < 8; ++v) {
            int row = rowBase + wave*16 + moff + v;
            L[(size_t)row*NC2PAD + col] = acc[nt][v] + bv;
        }
    }
}

// ---------------- softmax / argmax / segment epilogue ----------------
__global__ void k_rowfinal(const float* __restrict__ L, float* __restrict__ out,
                           float* __restrict__ s0buf) {
    int row = blockIdx.x * 256 + threadIdx.x;
    if (row >= NROWS) return;
    const float* lr = L + (size_t)row * NC2PAD;
    float m = -3.4e38f;
    #pragma unroll
    for (int c = 1; c <= 36; ++c) m = fmaxf(m, lr[c]);
    float e[36]; float sum = 0.f;
    #pragma unroll
    for (int j = 0; j < 36; ++j) { e[j] = expf(lr[j+1] - m); sum += e[j]; }
    float inv = 1.f / sum;
    #pragma unroll
    for (int j = 0; j < 36; ++j) { e[j] *= inv; out[(size_t)row*36 + j] = e[j]; }
    s0buf[row] = e[0];
    float best = e[1]; int bi = 1;
    #pragma unroll
    for (int j = 2; j < 36; ++j) if (e[j] > best) { best = e[j]; bi = j; }
    out[OUT_SCORES + row] = best;
    out[OUT_LABELS + row] = (float)(bi + 1);   // argmax over dist2[:,1:] + 2
}

__global__ void k_init_seg(int* segbits, unsigned int* cand) {
    int t = threadIdx.x;
    if (t < NSEG) { segbits[t] = 0; cand[t] = (unsigned)NROWS; }
}

__global__ void k_segmax(const float* __restrict__ s0, const int* __restrict__ bidx,
                         int* __restrict__ segbits) {
    int row = blockIdx.x * 256 + threadIdx.x;
    if (row >= NROWS) return;
    atomicMax(&segbits[bidx[row]], __float_as_int(s0[row]));   // s0 > 0 -> int order == float order
}

__global__ void k_candmin(const float* __restrict__ s0, const int* __restrict__ bidx,
                          const int* __restrict__ segbits, unsigned int* __restrict__ cand) {
    int row = blockIdx.x * 256 + threadIdx.x;
    if (row >= NROWS) return;
    int b = bidx[row];
    if (s0[row] >= __int_as_float(segbits[b])) atomicMin(&cand[b], (unsigned)row);
}

__global__ void k_apply_human(const unsigned int* __restrict__ cand,
                              const float* __restrict__ s0, float* __restrict__ out) {
    int t = threadIdx.x;
    if (t >= NSEG) return;
    unsigned idx = cand[t]; if (idx > NROWS - 1) idx = NROWS - 1;
    out[OUT_LABELS + idx] = 1.0f;
    out[OUT_SCORES + idx] = s0[idx];
}

// ---------------- launch ----------------
extern "C" void kernel_launch(void* const* d_in, const int* in_sizes, int n_in,
                              void* d_out, int out_size, void* d_ws, size_t ws_size,
                              hipStream_t stream) {
    const float* dist   = (const float*)d_in[0];
    const float* boxes  = (const float*)d_in[1];
    const float* feat   = (const float*)d_in[2];
    const int*   bidx   = (const int*)  d_in[3];
    const float* objW   = (const float*)d_in[4];
    const float* bn4g   = (const float*)d_in[5];
    const float* bn4b   = (const float*)d_in[6];
    const float* posW   = (const float*)d_in[7];
    const float* posB   = (const float*)d_in[8];
    const float* w1     = (const float*)d_in[9];
    const float* b1     = (const float*)d_in[10];
    const float* bn1g   = (const float*)d_in[11];
    const float* bn1b   = (const float*)d_in[12];
    const float* w2     = (const float*)d_in[13];
    const float* b2     = (const float*)d_in[14];
    float* out = (float*)d_out;

    char* p = (char*)d_ws;
    auto carve = [&](size_t bytes) -> void* {
        void* r = (void*)p;
        p += (bytes + 255) & ~(size_t)255;
        return r;
    };
    unsigned short* X    = (unsigned short*)carve((size_t)NROWS * KPAD * 2);
    unsigned short* W1T  = (unsigned short*)carve((size_t)H1 * KPAD * 2);
    unsigned short* W2T  = (unsigned short*)carve((size_t)NC2PAD * H1 * 2);
    float*          Y    = (float*)carve((size_t)NROWS * H1 * 4);
    unsigned short* H    = (unsigned short*)carve((size_t)NROWS * H1 * 2);
    float*          L    = (float*)carve((size_t)NROWS * NC2PAD * 4);
    float*          s0   = (float*)carve((size_t)NROWS * 4);
    float*          pSum = (float*)carve((size_t)64 * H1 * 4);
    float*          pSq  = (float*)carve((size_t)64 * H1 * 4);
    float*          bnSt = (float*)carve((size_t)2 * H1 * 4);
    float*          csSt = (float*)carve(8 * 4);
    int*            segb = (int*)carve(NSEG * 4);
    unsigned int*   cand = (unsigned int*)carve(NSEG * 4);

    k_cs_stats<<<1, 256, 0, stream>>>(boxes, csSt);
    k_cvt_w1<<<dim3(KPAD/32, H1/32), 256, 0, stream>>>(w1, W1T);
    k_cvt_w2<<<(NC2PAD*H1 + 255)/256, 256, 0, stream>>>(w2, W2T);
    k_build_x<<<NROWS/8, 256, 0, stream>>>(dist, boxes, feat, objW, posW, posB,
                                           bn4g, bn4b, csSt, X);
    k_gemm1<<<dim3(H1/128, NROWS/128), 256, 0, stream>>>(X, W1T, b1, Y);
    k_colstats<<<dim3(H1/256, 64), 256, 0, stream>>>(Y, pSum, pSq);
    k_colstats_final<<<H1/256, 256, 0, stream>>>(pSum, pSq, bnSt);
    k_norm_h<<<(NROWS*(H1/4))/256, 256, 0, stream>>>(Y, bnSt, bn1g, bn1b, H);
    k_gemm2<<<NROWS/128, 256, 0, stream>>>(H, W2T, b2, L);
    k_init_seg<<<1, 64, 0, stream>>>(segb, cand);
    k_rowfinal<<<NROWS/256, 256, 0, stream>>>(L, out, s0);
    k_segmax<<<NROWS/256, 256, 0, stream>>>(s0, bidx, segb);
    k_candmin<<<NROWS/256, 256, 0, stream>>>(s0, bidx, segb, cand);
    k_apply_human<<<1, 64, 0, stream>>>(cand, s0, out);

    (void)in_sizes; (void)n_in; (void)out_size; (void)ws_size;
}
